// SetPartitionGNN_17351667876280
// MI455X (gfx1250) — matrix operand order repaired and checked by reading, not compiled, not run on hardware
//
#include <hip/hip_runtime.h>
#include <hip/hip_bf16.h>

typedef __attribute__((ext_vector_type(8)))  float  v8f;
typedef __attribute__((ext_vector_type(8)))  __bf16 v8bf;
typedef __attribute__((ext_vector_type(16))) __bf16 v16bf;

#define BATCH 16
#define NNODE 2048
#define CIN   10
#define KNN_K 16
#define MTOT  (BATCH * NNODE)   // 32768

// ---------- helpers ----------
__device__ __forceinline__ __bf16 f2bf(float f) {
    unsigned u = __float_as_uint(f);
    unsigned r = (u + 0x7FFFu + ((u >> 16) & 1u)) >> 16;   // round-to-nearest-even
    unsigned short s = (unsigned short)r;
    __bf16 out;
    __builtin_memcpy(&out, &s, 2);
    return out;
}

// A-fragment (16x32 bf16): lane m = lane&15, 16 consecutive K values at kb = (lane>>4)*16
__device__ __forceinline__ v16bf load_afrag(const __bf16* p) {
    v8bf lo = *(const v8bf*)(p);
    v8bf hi = *(const v8bf*)(p + 8);
    v16bf r;
#pragma unroll
    for (int i = 0; i < 8; ++i) { r[i] = lo[i]; r[i + 8] = hi[i]; }
    return r;
}

// B-fragment (32x16 bf16): lane n = lane&15, 16 consecutive K values at koff = (lane>>4)*16
__device__ __forceinline__ v16bf load_bfrag(const __bf16* p) {
    v8bf lo = *(const v8bf*)(p);
    v8bf hi = *(const v8bf*)(p + 8);
    v16bf r;
#pragma unroll
    for (int i = 0; i < 8; ++i) { r[i] = lo[i]; r[i + 8] = hi[i]; }
    return r;
}

// ---------- kNN: one block = 256 rows of one batch, j tiled through LDS ----------
__global__ __launch_bounds__(256) void knn_kernel(const float* __restrict__ x,
                                                  int* __restrict__ nbr) {
    __shared__ float sx[256 * CIN];
    const int b = blockIdx.y;
    const int i = blockIdx.x * 256 + threadIdx.x;
    const float* xb = x + (size_t)b * NNODE * CIN;

    float xi[CIN];
#pragma unroll
    for (int c = 0; c < CIN; ++c) xi[c] = xb[i * CIN + c];

    float bestd[KNN_K];
    int   besti[KNN_K];
#pragma unroll
    for (int s = 0; s < KNN_K; ++s) { bestd[s] = 3.0e38f; besti[s] = 0; }

    for (int j0 = 0; j0 < NNODE; j0 += 256) {
        for (int t = threadIdx.x; t < 256 * CIN; t += 256) sx[t] = xb[j0 * CIN + t];
        __syncthreads();
        for (int jj = 0; jj < 256; ++jj) {
            const int j = j0 + jj;
            float d2 = 0.f;
#pragma unroll
            for (int c = 0; c < CIN; ++c) {
                float d = xi[c] - sx[jj * CIN + c];
                d2 = fmaf(d, d, d2);
            }
            if (j == i) d2 = 3.0e38f;
            if (d2 < bestd[KNN_K - 1]) {          // register-resident sorted insert
                float dd = d2;
                int ii = b * NNODE + j;           // flat node id (with batch offset)
#pragma unroll
                for (int s = 0; s < KNN_K; ++s) {
                    if (dd < bestd[s]) {
                        float td = bestd[s]; bestd[s] = dd; dd = td;
                        int ti = besti[s]; besti[s] = ii; ii = ti;
                    }
                }
            }
        }
        __syncthreads();
    }
#pragma unroll
    for (int s = 0; s < KNN_K; ++s)
        nbr[((size_t)b * NNODE + i) * KNN_K + s] = besti[s];
}

// ---------- pad x [*,10] -> h0 [*,32] fp32 ----------
__global__ void pad_x_kernel(const float* __restrict__ x, float* __restrict__ h0) {
    int tid = blockIdx.x * blockDim.x + threadIdx.x;
    if (tid >= MTOT * 32) return;
    int n = tid >> 5, c = tid & 31;
    h0[tid] = (c < CIN) ? x[n * CIN + c] : 0.f;
}

__global__ void zero_kernel(float* __restrict__ p, int n) {
    int tid = blockIdx.x * blockDim.x + threadIdx.x;
    if (tid < n) p[tid] = 0.f;
}

__global__ void conv_bf16_kernel(const float* __restrict__ src, __bf16* __restrict__ dst, int n) {
    int tid = blockIdx.x * blockDim.x + threadIdx.x;
    if (tid < n) dst[tid] = f2bf(src[tid]);
}

// W [Ktrue x N] fp32 -> Wt [N x Kpad] bf16 (transposed + zero-padded; B-fragment friendly)
__global__ void transw_kernel(const float* __restrict__ W, __bf16* __restrict__ Wt,
                              int Ktrue, int Kpad, int N) {
    int tid = blockIdx.x * blockDim.x + threadIdx.x;
    if (tid >= N * Kpad) return;
    int n = tid / Kpad, k = tid - n * Kpad;
    Wt[tid] = (k < Ktrue) ? f2bf(W[k * N + n]) : f2bf(0.f);
}

// ---------- scatter-add: agg[j] += h[i] for each j in knn(i) ----------
__global__ void scatter_kernel(const float* __restrict__ h, const int* __restrict__ nbr,
                               float* __restrict__ agg, int Kp) {
    int tid = blockIdx.x * blockDim.x + threadIdx.x;
    int i = tid / Kp;
    if (i >= MTOT) return;
    int c = tid - i * Kp;
    float v = h[(size_t)i * Kp + c];
    const int* nb = nbr + (size_t)i * KNN_K;
#pragma unroll
    for (int s = 0; s < KNN_K; ++s)
        atomicAdd(&agg[(size_t)nb[s] * Kp + c], v);   // GLOBAL_ATOMIC_ADD_F32
}

// ---------- dual GEMM: out = A1 @ W1 + A2 @ W2 + bias (bf16 WMMA, f32 acc) ----------
__global__ __launch_bounds__(256) void gemm_dual_kernel(
        const __bf16* __restrict__ A1, const __bf16* __restrict__ A2,
        const __bf16* __restrict__ W1t, const __bf16* __restrict__ W2t,
        const float* __restrict__ bias, float* __restrict__ out,
        int M, int Kpad, int N, int relu) {
    const int wave = (blockIdx.x * blockDim.x + threadIdx.x) >> 5;
    const int lane = threadIdx.x & 31;
    const int tilesN = N >> 4;
    const int mt = wave / tilesN, nt = wave - mt * tilesN;
    if (mt * 16 >= M) return;
    const int m0 = mt * 16, n0 = nt * 16;

    const int r    = lane & 15;            // A row / B col within tile
    const int kb   = (lane >> 4) * 16;     // A-fragment K sub-offset
    const int koff = (lane >> 4) * 16;     // B-fragment K sub-offset

    const __bf16* a1p = A1 + (size_t)(m0 + r) * Kpad + kb;
    const __bf16* a2p = A2 + (size_t)(m0 + r) * Kpad + kb;
    const __bf16* b1p = W1t + (size_t)(n0 + r) * Kpad + koff;
    const __bf16* b2p = W2t + (size_t)(n0 + r) * Kpad + koff;

    v8f c = {};
    for (int k0 = 0; k0 < Kpad; k0 += 32) {
        __builtin_prefetch(a1p + k0 + 32, 0, 3);
        v16bf a1 = load_afrag(a1p + k0);
        v16bf b1 = load_bfrag(b1p + k0);
        c = __builtin_amdgcn_wmma_f32_16x16x32_bf16(false, a1, false, b1,
                                                    (short)0, c, false, false);
        v16bf a2 = load_afrag(a2p + k0);
        v16bf b2 = load_bfrag(b2p + k0);
        c = __builtin_amdgcn_wmma_f32_16x16x32_bf16(false, a2, false, b2,
                                                    (short)0, c, false, false);
    }
    // C/D layout: VGPR r -> M = (lane>=16 ? 8 : 0) + r, N = lane&15
    const int nn = lane & 15;
    const int mb = (lane >> 4) * 8;
    const float bv = bias[n0 + nn];
#pragma unroll
    for (int rr = 0; rr < 8; ++rr) {
        float v = c[rr] + bv;
        if (relu) v = fmaxf(v, 0.f);
        out[(size_t)(m0 + mb + rr) * N + n0 + nn] = v;
    }
}

// ---------- final bilinear: out[b,i,j] = dot(h[b,i,:], h[b,j,:]), C=64 ----------
// Each wave computes a 16x64 output strip: A-fragment reused across 4 B-tiles
// (cuts L2 read amplification from 4x to 2.5x; 4 WMMA per K-loop iteration).
__global__ __launch_bounds__(256) void bilinear_kernel(const __bf16* __restrict__ hb,
                                                       float* __restrict__ out) {
    const int gwave = (blockIdx.x * blockDim.x + threadIdx.x) >> 5;
    const int lane  = threadIdx.x & 31;
    const int itiles  = NNODE >> 4;        // 128
    const int jstrips = NNODE >> 6;        // 32 strips of 64 cols
    const int perB  = itiles * jstrips;    // 4096
    const int b  = gwave / perB;
    const int t  = gwave - b * perB;
    const int it = t / jstrips, jt = t - it * jstrips;

    const __bf16* base = hb + (size_t)b * NNODE * 64;
    const int r    = lane & 15;
    const int kb   = (lane >> 4) * 16;
    const int koff = (lane >> 4) * 16;

    const __bf16* ap = base + (size_t)(it * 16 + r) * 64 + kb;
    const __bf16* bp = base + (size_t)(jt * 64 + r) * 64 + koff;   // +u*1024 per col-tile

    v8f c0 = {}, c1 = {}, c2 = {}, c3 = {};
#pragma unroll
    for (int k0 = 0; k0 < 64; k0 += 32) {
        v16bf a = load_afrag(ap + k0);
        v16bf b0 = load_bfrag(bp + 0 * 1024 + k0);
        c0 = __builtin_amdgcn_wmma_f32_16x16x32_bf16(false, a, false, b0,
                                                     (short)0, c0, false, false);
        v16bf b1 = load_bfrag(bp + 1 * 1024 + k0);
        c1 = __builtin_amdgcn_wmma_f32_16x16x32_bf16(false, a, false, b1,
                                                     (short)0, c1, false, false);
        v16bf b2 = load_bfrag(bp + 2 * 1024 + k0);
        c2 = __builtin_amdgcn_wmma_f32_16x16x32_bf16(false, a, false, b2,
                                                     (short)0, c2, false, false);
        v16bf b3 = load_bfrag(bp + 3 * 1024 + k0);
        c3 = __builtin_amdgcn_wmma_f32_16x16x32_bf16(false, a, false, b3,
                                                     (short)0, c3, false, false);
    }
    const int nn = lane & 15;
    const int mb = (lane >> 4) * 8;
    float* ob = out + (size_t)b * NNODE * NNODE + (size_t)(it * 16) * NNODE + jt * 64;
#pragma unroll
    for (int rr = 0; rr < 8; ++rr) {
        float* row = ob + (size_t)(mb + rr) * NNODE + nn;
        row[0]  = c0[rr];
        row[16] = c1[rr];
        row[32] = c2[rr];
        row[48] = c3[rr];
    }
}

// ---------- host ----------
extern "C" void kernel_launch(void* const* d_in, const int* in_sizes, int n_in,
                              void* d_out, int out_size, void* d_ws, size_t ws_size,
                              hipStream_t stream) {
    const float* x = (const float*)d_in[0];
    const float* wrel[4]  = {(const float*)d_in[1], (const float*)d_in[4],
                             (const float*)d_in[7], (const float*)d_in[10]};
    const float* wroot[4] = {(const float*)d_in[2], (const float*)d_in[5],
                             (const float*)d_in[8], (const float*)d_in[11]};
    const float* bias[4]  = {(const float*)d_in[3], (const float*)d_in[6],
                             (const float*)d_in[9], (const float*)d_in[12]};

    char* ws = (char*)d_ws;
    int*    nbr   = (int*)   (ws);                         //  2 MB
    float*  hA    = (float*) (ws + ((size_t)2  << 20));    // 16 MB
    float*  hB    = (float*) (ws + ((size_t)18 << 20));    // 16 MB
    float*  agg   = (float*) (ws + ((size_t)34 << 20));    // 16 MB
    __bf16* hbf   = (__bf16*)(ws + ((size_t)50 << 20));    //  8 MB
    __bf16* aggbf = (__bf16*)(ws + ((size_t)58 << 20));    //  8 MB
    __bf16* wt    = (__bf16*)(ws + ((size_t)66 << 20));    // ~176 KB

    __bf16* wtrel[4];
    __bf16* wtroot[4];
    const int Kpad[4]  = {32, 128, 128, 128};
    const int Ktrue[4] = {10, 128, 128, 128};
    const int Nout[4]  = {128, 128, 128, 64};
    {
        __bf16* p = wt;
        for (int li = 0; li < 4; ++li) {
            wtrel[li]  = p; p += (size_t)Nout[li] * Kpad[li];
            wtroot[li] = p; p += (size_t)Nout[li] * Kpad[li];
        }
    }

    // 1) kNN graph
    knn_kernel<<<dim3(NNODE / 256, BATCH), 256, 0, stream>>>(x, nbr);

    // 2) pad x -> h0 (fp32, stride 32)
    pad_x_kernel<<<(MTOT * 32 + 255) / 256, 256, 0, stream>>>(x, hA);

    // 3) weights -> bf16, transposed + padded
    for (int li = 0; li < 4; ++li) {
        int cnt = Nout[li] * Kpad[li];
        transw_kernel<<<(cnt + 255) / 256, 256, 0, stream>>>(wrel[li],  wtrel[li],  Ktrue[li], Kpad[li], Nout[li]);
        transw_kernel<<<(cnt + 255) / 256, 256, 0, stream>>>(wroot[li], wtroot[li], Ktrue[li], Kpad[li], Nout[li]);
    }

    // 4) GraphConv layers
    float* hcur = hA;
    float* hnext = hB;
    for (int li = 0; li < 4; ++li) {
        const int Kp = Kpad[li], Nn = Nout[li];
        const int elems = MTOT * Kp;

        conv_bf16_kernel<<<(elems + 255) / 256, 256, 0, stream>>>(hcur, hbf, elems);
        zero_kernel<<<(elems + 255) / 256, 256, 0, stream>>>(agg, elems);
        scatter_kernel<<<(elems + 255) / 256, 256, 0, stream>>>(hcur, nbr, agg, Kp);
        conv_bf16_kernel<<<(elems + 255) / 256, 256, 0, stream>>>(agg, aggbf, elems);

        const int waves  = (MTOT / 16) * (Nn / 16);
        const int blocks = waves / 8;                     // 8 waves / 256-thread block
        gemm_dual_kernel<<<blocks, 256, 0, stream>>>(aggbf, hbf, wtrel[li], wtroot[li],
                                                     bias[li], hnext, MTOT, Kp, Nn,
                                                     (li < 3) ? 1 : 0);
        float* tmp = hcur; hcur = hnext; hnext = tmp;
    }

    // 5) h (32768 x 64) -> bf16, then batched bilinear h h^T via WMMA (16x64 strips)
    conv_bf16_kernel<<<(MTOT * 64 + 255) / 256, 256, 0, stream>>>(hcur, hbf, MTOT * 64);
    {
        const int waves  = BATCH * (NNODE / 16) * (NNODE / 64);  // 65536
        const int blocks = waves / 8;                            // 8192
        bilinear_kernel<<<blocks, 256, 0, stream>>>(hbf, (float*)d_out);
    }
    (void)in_sizes; (void)n_in; (void)out_size; (void)ws_size;
}
